// PhoenixRanker_25769803776615
// MI455X (gfx1250) — matrix operand order repaired and checked by reading, not compiled, not run on hardware
//
#include <hip/hip_runtime.h>
#include <stdint.h>

// Problem constants
#define VV    100000
#define DD    256
#define HH    4
#define LL    4
#define DFF   1024
#define HIST  384
#define CANDN 128
#define BB    32
#define SS    512
#define HDIM  64
#define MROWS (BB * SS)   // 16384

typedef __bf16 bf16;
typedef __attribute__((ext_vector_type(16))) __bf16 v16bf;
typedef __attribute__((ext_vector_type(8)))  __bf16 v8bf;
typedef __attribute__((ext_vector_type(8)))  float  v8f;

// ---------------- helpers ----------------

__device__ inline unsigned short f2bf(float f) {
  union { float f; unsigned u; } c; c.f = f;
  unsigned u = c.u;
  u = u + 0x7FFFu + ((u >> 16) & 1u);   // round-to-nearest-even
  return (unsigned short)(u >> 16);
}

__device__ inline v8f wmma_bf16(v16bf a, v16bf b, v8f c) {
  return __builtin_amdgcn_wmma_f32_16x16x32_bf16(false, a, false, b, (short)0, c,
                                                 false, false);
}

__device__ inline v16bf make16(v8bf lo, v8bf hi) {
  v16bf r;
#pragma unroll
  for (int i = 0; i < 8; ++i) { r[i] = lo[i]; r[i + 8] = hi[i]; }
  return r;
}

// A-matrix fragment (16x32 bf16, M x K):
// lanes 0-15: row M=lane, K = {k0..k0+7} in V0-3 and {k0+16..k0+23} in V4-7
// lanes 16-31: same rows, K = {k0+8..15} and {k0+24..31}
__device__ inline v16bf load_a_frag(const unsigned short* A, int lda, int k0, int lane) {
  const unsigned short* p = A + (lane & 15) * lda + k0 + ((lane >> 4) << 3);
  v8bf lo = *(const v8bf*)(p);
  v8bf hi = *(const v8bf*)(p + 16);
  return make16(lo, hi);
}

// B-matrix fragment (32x16 bf16, K x N) from W stored (N,K) row-major:
// lanes 0-15: col N=lane, K = {k0..k0+7} and {k0+16..k0+23}; lanes 16-31: K = {k0+8..15} and {k0+24..31}
__device__ inline v16bf load_b_frag(const unsigned short* W, int ldk, int k0, int lane) {
  const unsigned short* p = W + (lane & 15) * ldk + k0 + ((lane >> 4) << 3);
  v8bf lo = *(const v8bf*)(p);
  v8bf hi = *(const v8bf*)(p + 16);
  return make16(lo, hi);
}

__device__ inline v8f vzero8() {
  v8f z;
#pragma unroll
  for (int i = 0; i < 8; ++i) z[i] = 0.f;
  return z;
}

// CDNA5 async copy: global -> LDS, 16B per lane, tracked by ASYNCcnt.
// GVS mode: SGPR 64-bit base + per-lane 32-bit byte offset; LDS dest per-lane.
__device__ inline void async_g2l_b128(uint32_t lds_byte_addr, uint32_t gvoff,
                                      uint64_t gbase) {
  asm volatile("global_load_async_to_lds_b128 %0, %1, %2"
               :: "v"(lds_byte_addr), "v"(gvoff), "s"(gbase) : "memory");
}

// ---------------- kernels ----------------

__global__ __launch_bounds__(256)
void f32_to_bf16_kernel(const float* __restrict__ src, unsigned short* __restrict__ dst, int n) {
  int i = blockIdx.x * 256 + threadIdx.x;
  if (i < n) dst[i] = f2bf(src[i]);
}

__global__ __launch_bounds__(256)
void embed_kernel(const int* __restrict__ hist_ids, const int* __restrict__ cand_ids,
                  const float* __restrict__ news, const float* __restrict__ pos,
                  float* __restrict__ x) {
  int bs = blockIdx.x;                // 0..MROWS-1
  int b = bs >> 9;                    // /SS
  int s = bs & (SS - 1);
  int tid = threadIdx.x;
  int id = (s < HIST) ? hist_ids[b * HIST + s] : cand_ids[b * CANDN + (s - HIST)];
  x[bs * DD + tid] = news[id * DD + tid] + pos[s * DD + tid];
}

__global__ __launch_bounds__(256)
void layernorm_bf16_kernel(const float* __restrict__ x, const float* __restrict__ g,
                           const float* __restrict__ bi, unsigned short* __restrict__ out) {
  __shared__ float red[256];
  int row = blockIdx.x, tid = threadIdx.x;
  float v = x[row * DD + tid];
  red[tid] = v; __syncthreads();
  for (int s = 128; s > 0; s >>= 1) { if (tid < s) red[tid] += red[tid + s]; __syncthreads(); }
  float mu = red[0] * (1.f / DD);
  __syncthreads();
  float d = v - mu;
  red[tid] = d * d; __syncthreads();
  for (int s = 128; s > 0; s >>= 1) { if (tid < s) red[tid] += red[tid + s]; __syncthreads(); }
  float rstd = rsqrtf(red[0] * (1.f / DD) + 1e-5f);
  out[row * DD + tid] = f2bf(d * rstd * g[tid] + bi[tid]);
}

// EPI: 0 = bf16 out, 1 = bf16 out + ReLU, 2 = f32 residual accumulate
// C[M,N] = A(MxK bf16, lda=K) * W(NxK bf16 row-major) + bias
// Weight tile (128 x 32) is staged in LDS via async copies, double-buffered:
// all 8 waves share it (4x less L2 traffic), s_wait_asynccnt pipelines the
// next tile's loads underneath the current tile's WMMAs.
#define WT_LDK 40   // padded LDS row stride in halves (32 data + 8 pad)

template <int EPI>
__global__ __launch_bounds__(256)
void gemm_bf16_kernel(const unsigned short* __restrict__ A,
                      const unsigned short* __restrict__ W,
                      const float* __restrict__ bias,
                      unsigned short* __restrict__ outb,
                      float* __restrict__ resid,
                      int N, int K) {
  __shared__ unsigned short wtile[2][128 * WT_LDK];   // 2 x 10KB

  const int tid   = threadIdx.x;
  const int lane  = tid & 31;
  const int wv    = tid >> 5;                          // 0..7 waves
  const int mbase = blockIdx.y * 128 + (wv >> 1) * 32;
  const int nwave = (wv & 1) * 64;                     // wave n-offset in block tile
  const int nblk  = blockIdx.x * 128;

  const uint64_t wg = (uint64_t)(uintptr_t)(W + (size_t)nblk * K);
  const int nk = K >> 5;

  // Per-thread slice of the 128x32 tile: 512 x 16B chunks, 2 per thread.
  const int r0 = (tid + 0)   >> 2, c0 = (tid + 0)   & 3;
  const int r1 = (tid + 256) >> 2, c1 = (tid + 256) & 3;

  // Prologue: stage k-tile 0.
  {
    uint32_t l0 = (uint32_t)(uintptr_t)&wtile[0][0];
    async_g2l_b128(l0 + r0 * (WT_LDK * 2) + c0 * 16, (uint32_t)((r0 * K + c0 * 8) * 2), wg);
    async_g2l_b128(l0 + r1 * (WT_LDK * 2) + c1 * 16, (uint32_t)((r1 * K + c1 * 8) * 2), wg);
  }

  v8f acc[2][4];
#pragma unroll
  for (int i = 0; i < 2; ++i)
#pragma unroll
    for (int j = 0; j < 4; ++j) acc[i][j] = vzero8();

  for (int k = 0; k < nk; ++k) {
    const int k0 = k << 5;
    if (k + 1 < nk) {
      // Kick off next tile into the other buffer, then wait for current tile
      // (allow the 2 just-issued ops to stay outstanding).
      uint32_t l1 = (uint32_t)(uintptr_t)&wtile[(k + 1) & 1][0];
      int kn = k0 + 32;
      async_g2l_b128(l1 + r0 * (WT_LDK * 2) + c0 * 16, (uint32_t)((r0 * K + kn + c0 * 8) * 2), wg);
      async_g2l_b128(l1 + r1 * (WT_LDK * 2) + c1 * 16, (uint32_t)((r1 * K + kn + c1 * 8) * 2), wg);
      asm volatile("s_wait_asynccnt 0x2" ::: "memory");
    } else {
      asm volatile("s_wait_asynccnt 0x0" ::: "memory");
    }
    __syncthreads();   // current tile visible from all waves' async writes

    const unsigned short* wt = &wtile[k & 1][0];
    v16bf a0 = load_a_frag(A + (size_t)(mbase +  0) * K, K, k0, lane);
    v16bf a1 = load_a_frag(A + (size_t)(mbase + 16) * K, K, k0, lane);
#pragma unroll
    for (int j = 0; j < 4; ++j) {
      v16bf b = load_b_frag(wt + (nwave + j * 16) * WT_LDK, WT_LDK, 0, lane);
      acc[0][j] = wmma_bf16(a0, b, acc[0][j]);
      acc[1][j] = wmma_bf16(a1, b, acc[1][j]);
    }
    __syncthreads();   // done reading this buffer before it is refilled
  }

  const int nlane = lane & 15;
  const int moff  = (lane >> 4) * 8;
#pragma unroll
  for (int i = 0; i < 2; ++i) {
#pragma unroll
    for (int j = 0; j < 4; ++j) {
      int col = nblk + nwave + j * 16 + nlane;
      float bv = bias[col];
#pragma unroll
      for (int v = 0; v < 8; ++v) {
        int row = mbase + i * 16 + moff + v;
        float val = acc[i][j][v] + bv;
        if (EPI == 1) val = val > 0.f ? val : 0.f;
        if (EPI == 2) resid[(size_t)row * N + col] += val;
        else          outb[(size_t)row * N + col] = f2bf(val);
      }
    }
  }
}

// Fused attention per (b, h, 32 query rows).
// qkv: (MROWS, 768) bf16; Q cols [0,256), K cols [256,512), V cols [512,768)
#define SSTR 516   // f32 score row stride (pad for LDS banks)
#define PSTR 528   // bf16 prob row stride (16B aligned)
#define VSTR 528   // bf16 V^T row stride

__global__ __launch_bounds__(256)
void attention_kernel(const unsigned short* __restrict__ qkv,
                      unsigned short* __restrict__ attn_out) {
  extern __shared__ unsigned char smem_raw[];
  float*          sc = (float*)smem_raw;                                  // 32 x SSTR
  unsigned short* pr = (unsigned short*)(smem_raw + 32 * SSTR * 4);       // 32 x PSTR
  unsigned short* vt = pr + 32 * PSTR;                                    // 64 x VSTR

  const int tid  = threadIdx.x;
  const int lane = tid & 31;
  const int wv   = tid >> 5;

  const int qblk = blockIdx.x & 15;
  const int h    = (blockIdx.x >> 4) & 3;
  const int b    = blockIdx.x >> 6;
  const int q0   = qblk * 32;
  const int base = b * SS;

  // V^T -> LDS (coalesced global reads, padded LDS rows)
  for (int idx = tid; idx < SS * HDIM; idx += 256) {
    int key = idx >> 6;
    int d   = idx & 63;
    vt[d * VSTR + key] = qkv[(size_t)(base + key) * 768 + 512 + h * 64 + d];
  }

  // ---- scores = Q K^T * scale + mask ----
  const unsigned short* Q  = qkv + (size_t)(base + q0) * 768 + h * 64;
  const unsigned short* Km = qkv + (size_t)base * 768 + 256 + h * 64;

  v16bf afr[2][2];
  afr[0][0] = load_a_frag(Q,            768, 0,  lane);
  afr[0][1] = load_a_frag(Q,            768, 32, lane);
  afr[1][0] = load_a_frag(Q + 16 * 768, 768, 0,  lane);
  afr[1][1] = load_a_frag(Q + 16 * 768, 768, 32, lane);

  const int nlane = lane & 15;
  const int moff  = (lane >> 4) * 8;

#pragma unroll
  for (int j = 0; j < 4; ++j) {
    int nt = wv * 4 + j;                               // key tile: 8 waves x 4 = 32
    v16bf b0 = load_b_frag(Km + (size_t)nt * 16 * 768, 768, 0,  lane);
    v16bf b1 = load_b_frag(Km + (size_t)nt * 16 * 768, 768, 32, lane);
#pragma unroll
    for (int mt = 0; mt < 2; ++mt) {
      v8f acc = vzero8();
      acc = wmma_bf16(afr[mt][0], b0, acc);
      acc = wmma_bf16(afr[mt][1], b1, acc);
      int col = nt * 16 + nlane;
#pragma unroll
      for (int v = 0; v < 8; ++v) {
        int rl   = mt * 16 + moff + v;                 // local query row 0..31
        int qrow = q0 + rl;
        float s  = acc[v] * 0.125f;                    // 1/sqrt(64)
        bool ok  = (col < HIST) || (col == qrow);      // isolation mask
        sc[rl * SSTR + col] = ok ? s : -__builtin_inff();
      }
    }
  }
  __syncthreads();

  // ---- softmax: 8 threads per row ----
  {
    int row = tid >> 3;
    int sub = tid & 7;
    float m = -__builtin_inff();
    for (int c = sub; c < SS; c += 8) m = fmaxf(m, sc[row * SSTR + c]);
    m = fmaxf(m, __shfl_xor(m, 1, 8));
    m = fmaxf(m, __shfl_xor(m, 2, 8));
    m = fmaxf(m, __shfl_xor(m, 4, 8));
    float sum = 0.f;
    for (int c = sub; c < SS; c += 8) {
      float e = __expf(sc[row * SSTR + c] - m);
      sc[row * SSTR + c] = e;
      sum += e;
    }
    sum += __shfl_xor(sum, 1, 8);
    sum += __shfl_xor(sum, 2, 8);
    sum += __shfl_xor(sum, 4, 8);
    float rinv = 1.f / sum;
    for (int c = sub; c < SS; c += 8)
      pr[row * PSTR + c] = f2bf(sc[row * SSTR + c] * rinv);
  }
  __syncthreads();

  // ---- out = P V : each wave one 16x16 tile (2 mtiles x 4 ntiles) ----
  {
    int mt = wv & 1;
    int nt = wv >> 1;
    v8f acc = vzero8();
    for (int k0 = 0; k0 < SS; k0 += 32) {
      v16bf a = load_a_frag(pr + mt * 16 * PSTR, PSTR, k0, lane);
      v16bf bfrag = load_b_frag(vt + nt * 16 * VSTR, VSTR, k0, lane);
      acc = wmma_bf16(a, bfrag, acc);
    }
    int col = h * 64 + nt * 16 + nlane;
#pragma unroll
    for (int v = 0; v < 8; ++v) {
      int row = base + q0 + mt * 16 + moff + v;
      attn_out[(size_t)row * DD + col] = f2bf(acc[v]);
    }
  }
}

// Final LN + 4 scalar heads over candidate rows; out shape (4, B, CAND)
__global__ __launch_bounds__(256)
void head_kernel(const float* __restrict__ x,
                 const float* __restrict__ g, const float* __restrict__ bi,
                 const float* __restrict__ w0, const float* __restrict__ b0,
                 const float* __restrict__ w1, const float* __restrict__ b1,
                 const float* __restrict__ w2, const float* __restrict__ b2,
                 const float* __restrict__ w3, const float* __restrict__ b3,
                 float* __restrict__ out) {
  __shared__ float red[256];
  int r = blockIdx.x;                 // 0 .. B*CAND-1
  int b = r / CANDN;
  int c = r - b * CANDN;
  int tid = threadIdx.x;
  int srow = b * SS + HIST + c;
  float v = x[srow * DD + tid];
  red[tid] = v; __syncthreads();
  for (int s = 128; s > 0; s >>= 1) { if (tid < s) red[tid] += red[tid + s]; __syncthreads(); }
  float mu = red[0] * (1.f / DD);
  __syncthreads();
  float d = v - mu;
  red[tid] = d * d; __syncthreads();
  for (int s = 128; s > 0; s >>= 1) { if (tid < s) red[tid] += red[tid + s]; __syncthreads(); }
  float hn = d * rsqrtf(red[0] * (1.f / DD) + 1e-5f) * g[tid] + bi[tid];

  const float* ws[4] = { w0, w1, w2, w3 };
  const float* bs[4] = { b0, b1, b2, b3 };
  for (int k = 0; k < 4; ++k) {
    __syncthreads();
    red[tid] = hn * ws[k][tid];
    __syncthreads();
    for (int s = 128; s > 0; s >>= 1) { if (tid < s) red[tid] += red[tid + s]; __syncthreads(); }
    if (tid == 0) out[k * (BB * CANDN) + r] = red[0] + bs[k][0];
  }
}

// ---------------- launch ----------------

extern "C" void kernel_launch(void* const* d_in, const int* in_sizes, int n_in,
                              void* d_out, int out_size, void* d_ws, size_t ws_size,
                              hipStream_t stream) {
  const int*   hist_ids = (const int*)d_in[0];
  const int*   cand_ids = (const int*)d_in[1];
  const float* news_emb = (const float*)d_in[2];
  const float* pos_emb  = (const float*)d_in[3];
  const float* ln1_g = (const float*)d_in[4];
  const float* ln1_b = (const float*)d_in[5];
  const float* in_w  = (const float*)d_in[6];
  const float* in_b  = (const float*)d_in[7];
  const float* out_w = (const float*)d_in[8];
  const float* out_b = (const float*)d_in[9];
  const float* ln2_g = (const float*)d_in[10];
  const float* ln2_b = (const float*)d_in[11];
  const float* ff1_w = (const float*)d_in[12];
  const float* ff1_b = (const float*)d_in[13];
  const float* ff2_w = (const float*)d_in[14];
  const float* ff2_b = (const float*)d_in[15];
  const float* lnf_g = (const float*)d_in[16];
  const float* lnf_b = (const float*)d_in[17];
  const float* click_w  = (const float*)d_in[18];
  const float* click_b  = (const float*)d_in[19];
  const float* like_w   = (const float*)d_in[20];
  const float* like_b   = (const float*)d_in[21];
  const float* reply_w  = (const float*)d_in[22];
  const float* reply_b  = (const float*)d_in[23];
  const float* repost_w = (const float*)d_in[24];
  const float* repost_b = (const float*)d_in[25];
  float* out = (float*)d_out;

  uint8_t* wsb = (uint8_t*)d_ws;
  size_t off = 0;
  auto walloc = [&](size_t bytes) -> void* {
    void* p = wsb + off;
    off = (off + bytes + 255) & ~(size_t)255;
    return p;
  };
  float*          x    = (float*)walloc((size_t)MROWS * DD * 4);
  unsigned short* bufA = (unsigned short*)walloc((size_t)MROWS * DD * 2);
  unsigned short* bufB = (unsigned short*)walloc((size_t)MROWS * DFF * 2);
  unsigned short* w_in = (unsigned short*)walloc((size_t)LL * 3 * DD * DD * 2);
  unsigned short* w_o  = (unsigned short*)walloc((size_t)LL * DD * DD * 2);
  unsigned short* w_f1 = (unsigned short*)walloc((size_t)LL * DFF * DD * 2);
  unsigned short* w_f2 = (unsigned short*)walloc((size_t)LL * DD * DFF * 2);

  auto conv = [&](const float* s, unsigned short* d, int n) {
    f32_to_bf16_kernel<<<(n + 255) / 256, 256, 0, stream>>>(s, d, n);
  };
  conv(in_w,  w_in, LL * 3 * DD * DD);
  conv(out_w, w_o,  LL * DD * DD);
  conv(ff1_w, w_f1, LL * DFF * DD);
  conv(ff2_w, w_f2, LL * DD * DFF);

  embed_kernel<<<MROWS, 256, 0, stream>>>(hist_ids, cand_ids, news_emb, pos_emb, x);

  const size_t att_lds = 32 * SSTR * 4 + 32 * PSTR * 2 + 64 * VSTR * 2;  // ~164 KB

  for (int l = 0; l < LL; ++l) {
    layernorm_bf16_kernel<<<MROWS, 256, 0, stream>>>(x, ln1_g + l * DD, ln1_b + l * DD, bufA);
    gemm_bf16_kernel<0><<<dim3((3 * DD) / 128, MROWS / 128), 256, 0, stream>>>(
        bufA, w_in + (size_t)l * 3 * DD * DD, in_b + l * 3 * DD, bufB, nullptr, 3 * DD, DD);
    attention_kernel<<<BB * HH * (SS / 32), 256, att_lds, stream>>>(bufB, bufA);
    gemm_bf16_kernel<2><<<dim3(DD / 128, MROWS / 128), 256, 0, stream>>>(
        bufA, w_o + (size_t)l * DD * DD, out_b + l * DD, nullptr, x, DD, DD);
    layernorm_bf16_kernel<<<MROWS, 256, 0, stream>>>(x, ln2_g + l * DD, ln2_b + l * DD, bufA);
    gemm_bf16_kernel<1><<<dim3(DFF / 128, MROWS / 128), 256, 0, stream>>>(
        bufA, w_f1 + (size_t)l * DFF * DD, ff1_b + l * DFF, bufB, nullptr, DFF, DD);
    gemm_bf16_kernel<2><<<dim3(DD / 128, MROWS / 128), 256, 0, stream>>>(
        bufB, w_f2 + (size_t)l * DD * DFF, ff2_b + l * DD, nullptr, x, DD, DFF);
  }

  head_kernel<<<BB * CANDN, 256, 0, stream>>>(x, lnf_g, lnf_b,
                                              click_w, click_b, like_w, like_b,
                                              reply_w, reply_b, repost_w, repost_b, out);
}
